// StablelmAttention_11510512353528
// MI455X (gfx1250) — compile-verified, matrix operand order checked by reading, not run on hardware
//
#include <hip/hip_runtime.h>
#include <hip/hip_bf16.h>

// ---------------------------------------------------------------------------
// Types for CDNA5 WMMA (wave32): 16x16x32 bf16 -> f32
// ---------------------------------------------------------------------------
typedef __bf16 bf16x8  __attribute__((ext_vector_type(8)));
typedef __bf16 v16bf   __attribute__((ext_vector_type(16)));
typedef float  v8f     __attribute__((ext_vector_type(8)));
typedef unsigned int u32x4 __attribute__((ext_vector_type(4)));
typedef int          i32x8 __attribute__((ext_vector_type(8)));
typedef int          i32x4 __attribute__((ext_vector_type(4)));

__device__ __forceinline__ v8f wmma_bf16(v16bf a, v16bf b, v8f c) {
    return __builtin_amdgcn_wmma_f32_16x16x32_bf16(
        /*neg_a=*/false, a, /*neg_b=*/false, b,
        /*c_mod=*/(short)0, c, /*reuse_a=*/false, /*reuse_b=*/false);
}

// Load one lane's 16 bf16 values of an A-fragment (16x32, row per lane) or a
// B-fragment (32x16, column per lane) from a row-major buffer.
// rowptr points at element (row, k0). Per ISA packing, lane needs K runs
// [hf*8, hf*8+8) and [16+hf*8, 16+hf*8+8).
__device__ __forceinline__ v16bf load_frag(const __bf16* rowptr, int hf) {
    bf16x8 lo = *(const bf16x8*)(rowptr + hf * 8);
    bf16x8 hi = *(const bf16x8*)(rowptr + 16 + hf * 8);
    v16bf r;
#pragma unroll
    for (int i = 0; i < 8; ++i) { r[i] = lo[i]; r[i + 8] = hi[i]; }
    return r;
}

// ---------------------------------------------------------------------------
// Tensor Data Mover: DMA a 2D tile (tile_w x tile_h bf16 elements, row stride
// row_stride elements) from global memory into LDS, padding each row by
// (pad_amount_code+1) DWORDs every 2^(pad_interval_code+1) DWORDs.
// D# packed per CDNA5 ISA 8.3/8.4; groups 2/3 unused (2D tensor).
// This toolchain's builtin takes 6 operands (g0, g1, g2, g3, g4-spare, cpol).
// ---------------------------------------------------------------------------
__device__ __forceinline__ void tdm_load_2d(const void* gptr, void* ldsptr,
                                            unsigned tile_w, unsigned tile_h,
                                            unsigned long long row_stride,
                                            unsigned pad_interval_code,
                                            unsigned pad_amount_code) {
    const unsigned long long ga = (unsigned long long)(uintptr_t)gptr;
    const unsigned la = (unsigned)(uintptr_t)ldsptr;   // LDS byte offset

    u32x4 g0;
    g0[0] = 1u;                                   // count=1 (valid), flags 0
    g0[1] = la;                                   // lds_addr
    g0[2] = (unsigned)ga;                         // global_addr[31:0]
    g0[3] = (unsigned)((ga >> 32) & 0x01FFFFFFull) | (2u << 30);  // addr|type=2

    i32x8 g1;
    g1[0] = (int)((1u << 16) |                    // data_size = 1 (2 bytes)
                  (1u << 20) |                    // pad_enable
                  (pad_interval_code << 22) |
                  (pad_amount_code << 25));
    g1[1] = (int)((tile_w & 0xFFFFu) << 16);      // tensor_dim0[15:0]
    g1[2] = (int)((tile_h & 0xFFFFu) << 16);      // dim0 hi=0 | tensor_dim1 lo
    g1[3] = (int)(tile_w << 16);                  // dim1 hi=0 | tile_dim0
    g1[4] = (int)tile_h;                          // tile_dim1 | tile_dim2=0
    g1[5] = (int)(unsigned)row_stride;            // tensor_dim0_stride[31:0]
    g1[6] = (int)((unsigned)(row_stride >> 32) & 0xFFFFu);  // stride hi
    g1[7] = 0;

    const i32x4 z4 = {0, 0, 0, 0};
    const i32x8 z8 = {0, 0, 0, 0, 0, 0, 0, 0};
    __builtin_amdgcn_tensor_load_to_lds(g0, g1, z4, z4, z8, 0);
}

// ---------------------------------------------------------------------------
// Problem constants
// ---------------------------------------------------------------------------
constexpr int Bsz = 2, S = 2048, H = 2048, NH = 32, HD = 64, ROT = 16;
constexpr int Mrows = Bsz * S;        // 4096
constexpr int QKVN  = 3 * H;          // 6144

// ---------------------------------------------------------------------------
// fp32 -> bf16 conversion (vectorized, sizes are multiples of 4)
// ---------------------------------------------------------------------------
__global__ __launch_bounds__(256) void cvt_f32_to_bf16(
    const float* __restrict__ in, __bf16* __restrict__ out, int n) {
    int i = (blockIdx.x * 256 + threadIdx.x) * 4;
    if (i + 3 < n) {
        float4 f = *(const float4*)(in + i);
        out[i + 0] = (__bf16)f.x;
        out[i + 1] = (__bf16)f.y;
        out[i + 2] = (__bf16)f.z;
        out[i + 3] = (__bf16)f.w;
    }
}

// ---------------------------------------------------------------------------
// GEMM: C[MxN] = A[MxK] * B[KxN], all bf16 row-major in, bf16 or f32 out.
// Block tile 128x128x32; 8 waves, each wave computes 64x32 = 4x2 WMMA tiles.
// A tile staged by the Tensor Data Mover (with row padding to LDP pitch);
// B tile staged transposed by the wavefronts.
// ---------------------------------------------------------------------------
constexpr int BM = 128, BN = 128, BK = 32, LDP = BK + 8;   // pitch 40 (80B)

template <bool OUT_F32>
__global__ __launch_bounds__(256) void gemm_bf16_kernel(
    const __bf16* __restrict__ A, const __bf16* __restrict__ Bw,
    float* __restrict__ Cf, __bf16* __restrict__ Cb,
    int M, int N, int K) {
    __shared__ __bf16 sA[BM][LDP];
    __shared__ __bf16 sBt[BN][LDP];   // B staged transposed: sBt[n][k]

    const int tid  = threadIdx.x;
    const int wave = tid >> 5;
    const int lane = tid & 31;
    const int hf   = lane >> 4;
    const int lrow = lane & 15;
    const int wr   = wave >> 2;   // 0..1 : M position
    const int wc   = wave & 3;    // 0..3 : N position

    const int m0 = blockIdx.y * BM;
    const int n0 = blockIdx.x * BN;

    v8f acc[4][2];
#pragma unroll
    for (int mi = 0; mi < 4; ++mi)
#pragma unroll
        for (int ni = 0; ni < 2; ++ni)
#pragma unroll
            for (int r = 0; r < 8; ++r) acc[mi][ni][r] = 0.0f;

    const int brow = tid >> 3;           // 0..31
    const int bnc  = (tid & 7) * 16;     // 0..112

    for (int k0 = 0; k0 < K; k0 += BK) {
        // --- A tile 128x32 via TDM: row = 16 DWORDs, pad 4 DWORDs -> pitch 40
        if (wave == 0) {
            tdm_load_2d(A + (size_t)m0 * K + k0, &sA[0][0],
                        /*w=*/BK, /*h=*/BM, /*stride=*/(unsigned long long)K,
                        /*pad_interval=*/3, /*pad_amount=*/3);
            __builtin_amdgcn_s_wait_tensorcnt(0);
        }

        // --- B tile 32x128, staged transposed by all 256 threads
        const __bf16* gB = Bw + (size_t)(k0 + brow) * N + n0 + bnc;
        bf16x8 b0 = *(const bf16x8*)(gB);
        bf16x8 b1 = *(const bf16x8*)(gB + 8);
#pragma unroll
        for (int j = 0; j < 8; ++j) {
            sBt[bnc + j][brow]     = b0[j];
            sBt[bnc + 8 + j][brow] = b1[j];
        }
        if (k0 + BK < K)
            __builtin_prefetch(gB + (size_t)BK * N, 0, 1);   // global_prefetch
        __syncthreads();

        v16bf af[4], bfg[2];
#pragma unroll
        for (int mi = 0; mi < 4; ++mi)
            af[mi] = load_frag(&sA[wr * 64 + mi * 16 + lrow][0], hf);
#pragma unroll
        for (int ni = 0; ni < 2; ++ni)
            bfg[ni] = load_frag(&sBt[wc * 32 + ni * 16 + lrow][0], hf);
#pragma unroll
        for (int mi = 0; mi < 4; ++mi)
#pragma unroll
            for (int ni = 0; ni < 2; ++ni)
                acc[mi][ni] = wmma_bf16(af[mi], bfg[ni], acc[mi][ni]);
        __syncthreads();
    }

    // Epilogue: C layout — lane holds (M = r + 8*hf, N = lrow) per VGPR r.
#pragma unroll
    for (int mi = 0; mi < 4; ++mi)
#pragma unroll
        for (int ni = 0; ni < 2; ++ni) {
            const int col = n0 + wc * 32 + ni * 16 + lrow;
#pragma unroll
            for (int r = 0; r < 8; ++r) {
                const int row = m0 + wr * 64 + mi * 16 + r + 8 * hf;
                if constexpr (OUT_F32)
                    Cf[(size_t)row * N + col] = acc[mi][ni][r];
                else
                    Cb[(size_t)row * N + col] = (__bf16)acc[mi][ni][r];
            }
        }
}

// ---------------------------------------------------------------------------
// Partial RoPE on Q and K (first ROT=16 dims of each head), in-place on bf16
// qkv buffer [Mrows][6144]. One thread per (row, head, pair, q|k).
// ---------------------------------------------------------------------------
__global__ __launch_bounds__(256) void rope_kernel(
    const int* __restrict__ positions, __bf16* __restrict__ qkv) {
    const int idx  = blockIdx.x * 256 + threadIdx.x;   // Mrows*NH*8*2 threads
    const int pair = idx & 7;
    const int head = (idx >> 3) & 31;
    const int qk   = (idx >> 8) & 1;
    const int row  = idx >> 9;
    if (row >= Mrows) return;

    const int s = row & (S - 1);
    const float pos = (float)positions[s];
    const float expo = -(float)(2 * pair) / (float)ROT;
    const float inv_freq = __powf(10000.0f, expo);
    const float fr = pos * inv_freq;
    const float c = __cosf(fr), sn = __sinf(fr);

    __bf16* p = qkv + (size_t)row * QKVN + qk * H + head * HD;
    const float x1 = (float)p[pair];
    const float x2 = (float)p[pair + 8];
    p[pair]     = (__bf16)(x1 * c - x2 * sn);
    p[pair + 8] = (__bf16)(x2 * c + x1 * sn);
}

// ---------------------------------------------------------------------------
// Flash attention: one workgroup = (b, h, 64 q-rows); 4 waves x 16 q-rows.
// K tile staged by the TDM (rows are QK^T B-fragments), V staged transposed
// by the waves. Online softmax; P re-laid out via per-wave LDS tile for PV.
// ---------------------------------------------------------------------------
__global__ __launch_bounds__(128) void flash_attn_kernel(
    const __bf16* __restrict__ qkv, __bf16* __restrict__ attn) {
    __shared__ __bf16 sK[64][72];
    __shared__ __bf16 sVt[64][72];
    __shared__ __bf16 sP[4][16][72];

    const int qt = blockIdx.x & 31;          // q tile (S/64 = 32)
    const int h  = (blockIdx.x >> 5) & 31;   // head
    const int b  = blockIdx.x >> 10;         // batch

    const int tid = threadIdx.x, wave = tid >> 5, lane = tid & 31;
    const int hf = lane >> 4, lrow = lane & 15;

    // Q fragments for this wave's 16 rows (A layout: lane = row)
    const int qrow = qt * 64 + wave * 16 + lrow;
    const __bf16* qrp = qkv + (size_t)(b * S + qrow) * QKVN + h * HD;
    const v16bf aQ0 = load_frag(qrp, hf);        // d = 0..31
    const v16bf aQ1 = load_frag(qrp + 32, hf);   // d = 32..63

    float m_i[8], l_i[8];
    v8f oAcc[4];
#pragma unroll
    for (int r = 0; r < 8; ++r) { m_i[r] = -3.0e38f; l_i[r] = 0.0f; }
#pragma unroll
    for (int nt = 0; nt < 4; ++nt)
#pragma unroll
        for (int r = 0; r < 8; ++r) oAcc[nt][r] = 0.0f;

    const int kvrow = tid >> 1;          // 0..63
    const int kvc   = (tid & 1) * 32;    // 0 or 32
    const float scale = 0.125f;          // 1/sqrt(64)

    const int nkt = qt + 1;              // causal: only tiles kb <= qb
    for (int kt = 0; kt < nkt; ++kt) {
        const int kb = kt * 64;

        // --- K tile 64x64 via TDM: row = 32 DWORDs, pad 4 DWORDs -> pitch 72
        if (wave == 0) {
            tdm_load_2d(qkv + (size_t)(b * S + kb) * QKVN + H + h * HD,
                        &sK[0][0], /*w=*/64, /*h=*/64,
                        /*stride=*/(unsigned long long)QKVN,
                        /*pad_interval=*/4, /*pad_amount=*/3);
            __builtin_amdgcn_s_wait_tensorcnt(0);
        }

        // --- V tile 64x64, staged transposed by all 128 threads
        const __bf16* gV =
            qkv + (size_t)(b * S + kb + kvrow) * QKVN + 2 * H + h * HD + kvc;
        bf16x8 vv[4];
#pragma unroll
        for (int j = 0; j < 4; ++j) vv[j] = *(const bf16x8*)(gV + j * 8);
#pragma unroll
        for (int j = 0; j < 32; ++j) sVt[kvc + j][kvrow] = vv[j >> 3][j & 7];
        __syncthreads();

        // Scores: 16(q) x 64(k) via 4 column tiles, K-depth 64 = 2 WMMAs.
        v8f sc[4];
#pragma unroll
        for (int nt = 0; nt < 4; ++nt) {
#pragma unroll
            for (int r = 0; r < 8; ++r) sc[nt][r] = 0.0f;
            v16bf bK0 = load_frag(&sK[nt * 16 + lrow][0], hf);
            v16bf bK1 = load_frag(&sK[nt * 16 + lrow][32], hf);
            sc[nt] = wmma_bf16(aQ0, bK0, sc[nt]);
            sc[nt] = wmma_bf16(aQ1, bK1, sc[nt]);
        }

        // Scale + causal mask (only diagonal tile can violate causality)
#pragma unroll
        for (int nt = 0; nt < 4; ++nt)
#pragma unroll
            for (int r = 0; r < 8; ++r) {
                float v = sc[nt][r] * scale;
                if (kt == qt) {
                    const int kg = kb + nt * 16 + lrow;
                    const int qg = qt * 64 + wave * 16 + r + 8 * hf;
                    if (kg > qg) v = -3.0e38f;
                }
                sc[nt][r] = v;
            }

        // Online softmax, per q-row r (row lives on all 16 lanes of a half)
#pragma unroll
        for (int r = 0; r < 8; ++r) {
            float mx = fmaxf(fmaxf(sc[0][r], sc[1][r]), fmaxf(sc[2][r], sc[3][r]));
#pragma unroll
            for (int off = 8; off >= 1; off >>= 1)
                mx = fmaxf(mx, __shfl_xor(mx, off, 32));
            const float mnew  = fmaxf(m_i[r], mx);
            const float alpha = __expf(m_i[r] - mnew);
            m_i[r] = mnew;
            float rsum = 0.0f;
#pragma unroll
            for (int nt = 0; nt < 4; ++nt) {
                const float p = __expf(sc[nt][r] - mnew);
                sc[nt][r] = p;
                rsum += p;
            }
#pragma unroll
            for (int off = 8; off >= 1; off >>= 1)
                rsum += __shfl_xor(rsum, off, 32);
            l_i[r] = l_i[r] * alpha + rsum;
#pragma unroll
            for (int nt = 0; nt < 4; ++nt) oAcc[nt][r] *= alpha;
            // stash P (C layout -> row-major LDS) for re-read as A fragments
#pragma unroll
            for (int nt = 0; nt < 4; ++nt)
                sP[wave][r + 8 * hf][nt * 16 + lrow] = (__bf16)sc[nt][r];
        }

        // O += P(16x64) * V(64x64); per-wave sP, same-wave LDS is ordered.
        const v16bf aP0 = load_frag(&sP[wave][lrow][0], hf);
        const v16bf aP1 = load_frag(&sP[wave][lrow][32], hf);
#pragma unroll
        for (int nt = 0; nt < 4; ++nt) {
            v16bf bV0 = load_frag(&sVt[nt * 16 + lrow][0], hf);
            v16bf bV1 = load_frag(&sVt[nt * 16 + lrow][32], hf);
            oAcc[nt] = wmma_bf16(aP0, bV0, oAcc[nt]);
            oAcc[nt] = wmma_bf16(aP1, bV1, oAcc[nt]);
        }
        __syncthreads();   // before next tile overwrites sK / sVt
    }

    // Normalize and write attn output (bf16, [Mrows][H], col = h*64 + d)
#pragma unroll
    for (int nt = 0; nt < 4; ++nt) {
        const int col = h * HD + nt * 16 + lrow;
#pragma unroll
        for (int r = 0; r < 8; ++r) {
            const int row = b * S + qt * 64 + wave * 16 + r + 8 * hf;
            attn[(size_t)row * H + col] = (__bf16)(oAcc[nt][r] / l_i[r]);
        }
    }
}

// ---------------------------------------------------------------------------
// Host launcher
// ---------------------------------------------------------------------------
extern "C" void kernel_launch(void* const* d_in, const int* in_sizes, int n_in,
                              void* d_out, int out_size, void* d_ws, size_t ws_size,
                              hipStream_t stream) {
    const int*   positions = (const int*)d_in[0];
    const float* hidden    = (const float*)d_in[1];
    const float* wqkv      = (const float*)d_in[2];
    const float* wo        = (const float*)d_in[3];
    float*       out       = (float*)d_out;

    char* ws = (char*)d_ws;
    __bf16* Xb    = (__bf16*)ws;  ws += (size_t)Mrows * H * 2;       // 16 MB
    __bf16* Wqkvb = (__bf16*)ws;  ws += (size_t)H * QKVN * 2;        // 24 MB
    __bf16* Wob   = (__bf16*)ws;  ws += (size_t)H * H * 2;           //  8 MB
    __bf16* QKVb  = (__bf16*)ws;  ws += (size_t)Mrows * QKVN * 2;    // 48 MB
    __bf16* Attnb = (__bf16*)ws;                                      // 16 MB

    const int n1 = Mrows * H, n2 = H * QKVN, n3 = H * H;
    cvt_f32_to_bf16<<<n1 / 1024, 256, 0, stream>>>(hidden, Xb, n1);
    cvt_f32_to_bf16<<<n2 / 1024, 256, 0, stream>>>(wqkv, Wqkvb, n2);
    cvt_f32_to_bf16<<<n3 / 1024, 256, 0, stream>>>(wo, Wob, n3);

    // QKV projection: [4096 x 2048] * [2048 x 6144] -> bf16
    gemm_bf16_kernel<false><<<dim3(QKVN / BN, Mrows / BM), 256, 0, stream>>>(
        Xb, Wqkvb, nullptr, QKVb, Mrows, QKVN, H);

    // Partial RoPE on Q and K
    const int nrope = Mrows * NH * 8 * 2;   // 2,097,152
    rope_kernel<<<nrope / 256, 256, 0, stream>>>(positions, QKVb);

    // Fused causal flash attention: 2 * 32 * 32 workgroups
    flash_attn_kernel<<<Bsz * NH * (S / 64), 128, 0, stream>>>(QKVb, Attnb);

    // Output projection: [4096 x 2048] * [2048 x 2048] -> fp32 d_out
    gemm_bf16_kernel<true><<<dim3(H / BN, Mrows / BM), 256, 0, stream>>>(
        Attnb, Wob, out, nullptr, Mrows, H, H);
}
